// FARGAN_7541962571947
// MI455X (gfx1250) — matrix operand order repaired and checked by reading, not compiled, not run on hardware
//
#include <hip/hip_runtime.h>
#include <hip/hip_bf16.h>

// ---------------------------------------------------------------------------
// FARGAN vocoder recurrence on MI455X (gfx1250).
// 16 persistent workgroups; each owns an N-shard of every per-subframe weight
// matrix resident in LDS (~280KB/WGP of the 320KB), so the 400-step dependent
// chain never re-streams weights. Weight shards are staged into LDS with the
// Tensor Data Mover (TENSOR_LOAD_TO_LDS + s_wait_tensorcnt). All matmuls run
// on v_wmma_f32_16x16x32_bf16 (B=64 -> 4 M-tiles). Activations exchanged
// through L2 with a grid barrier (+ s_cluster_barrier, the hardware path under
// a 16-WG cluster dispatch and an architected NOP otherwise).
// ---------------------------------------------------------------------------

typedef __attribute__((ext_vector_type(16))) __bf16   v16bf;
typedef __attribute__((ext_vector_type(8)))  float    v8f;
typedef __attribute__((ext_vector_type(4)))  unsigned u32x4;
typedef __attribute__((ext_vector_type(4)))  int      i32x4;
typedef __attribute__((ext_vector_type(8)))  int      i32x8;

union BF16Frag { v16bf v; u32x4 q[2]; };

// ---------------- sizes -----------------------------------------------------
constexpr int B      = 64;     // batch
constexpr int FRAMES = 100;
constexpr int HOP    = 256;
constexpr int SUB    = 64;
constexpr int KC     = 352;    // cond K (336) padded to mult of 32

#define AL256(x) ((((size_t)(x)) + 255) & ~(size_t)255)

// ---------------- LDS layout -------------------------------------------------
constexpr int YL_ELEMS  = 64 * 48;                 // fp32 staging tile
constexpr int LDS_WBYTE = 2 * YL_ELEMS * 4;        // after Ylx,Ylh
// bf16-element offsets inside weight region (per-WG shards):
constexpr int LW_FW    = 0;                        // 16 x 528
constexpr int LW_FWGLU = LW_FW    + 16 * 528;      // 16 x 256
constexpr int LW_WIH0  = LW_FWGLU + 16 * 256;      // 48 x 384 each
constexpr int LW_WIH1  = LW_WIH0  + 48 * 384;
constexpr int LW_WIH2  = LW_WIH1  + 48 * 384;
constexpr int LW_WHH0  = LW_WIH2  + 48 * 384;      // 48 x 256 each
constexpr int LW_WHH1  = LW_WHH0  + 48 * 256;
constexpr int LW_WHH2  = LW_WHH1  + 48 * 256;
constexpr int LW_GLU0  = LW_WHH2  + 48 * 256;      // 16 x 256 each
constexpr int LW_GLU1  = LW_GLU0  + 16 * 256;
constexpr int LW_GLU2  = LW_GLU1  + 16 * 256;
constexpr int LW_SKIP  = LW_GLU2  + 16 * 256;      // 16 x 1152
constexpr int LW_SKGLU = LW_SKIP  + 16 * 1152;     // 16 x 256
constexpr int LW_OUT   = LW_SKGLU + 16 * 256;      // 16 x 256
constexpr int LW_END   = LW_OUT   + 16 * 256;
constexpr int LDS_BYTES = LDS_WBYTE + LW_END * 2;  // 311,808 B < 320KB WGP LDS

// ---------------- workspace layout (bytes) ----------------------------------
constexpr size_t WS_SYNC = 0;                                   // u32 barrier counter
constexpr size_t WS_CW1  = 256;                                 // bf16 [336][352]
constexpr size_t WS_CW2  = WS_CW1 + (size_t)336 * KC * 2;
constexpr size_t WS_CW3  = WS_CW2 + (size_t)336 * KC * 2;       // bf16 [512][352]
constexpr size_t WS_ACIN = AL256(WS_CW3 + (size_t)512 * KC * 2);// bf16 [64][352]
constexpr size_t WS_AC1  = WS_ACIN + (size_t)B * KC * 2;
constexpr size_t WS_AC2  = WS_AC1  + (size_t)B * KC * 2;
constexpr size_t WS_COND = WS_AC2  + (size_t)B * KC * 2;        // bf16 [64][512]
constexpr size_t WS_X4   = WS_COND + (size_t)B * 512 * 2;       // bf16 [64][528]
constexpr size_t WS_FW   = WS_X4   + (size_t)B * 528 * 2;       // bf16 [64][256] (fw0 / sk0)
constexpr size_t WS_SK   = WS_FW   + (size_t)B * 256 * 2;       // bf16 [64][256]
constexpr size_t WS_XG   = WS_SK   + (size_t)B * 256 * 2;       // bf16 [3][64][384]
constexpr size_t WS_H    = WS_XG   + (size_t)3 * B * 384 * 2;   // bf16 [2 parity][3][64][256]
constexpr size_t WS_SKX  = WS_H    + (size_t)2 * 3 * B * 256 * 2;// bf16 [64][1152]
constexpr size_t WS_HF   = AL256(WS_SKX + (size_t)B * 1152 * 2);// f32 [3][64][256]
constexpr size_t WS_S4   = WS_HF   + (size_t)3 * B * 256 * 4;   // f32 [64][264]
constexpr size_t WS_RING = WS_S4   + (size_t)B * 264 * 4;       // f32 [64][256] sample ring
constexpr size_t WS_GAIN = WS_RING + (size_t)B * 256 * 4;       // f32 [64]
constexpr size_t WS_PG   = WS_GAIN + (size_t)B * 4;             // f32 [64][4]
constexpr size_t WS_PL   = WS_PG   + (size_t)B * 4 * 4;         // f32 [64][72]
constexpr size_t WS_PSUB = WS_PL   + (size_t)B * 72 * 4;        // f32 [64][64]
constexpr size_t WS_WIMG = AL256(WS_PSUB + (size_t)B * 64 * 4); // bf16 [16 WG][LW_END] LDS images
constexpr size_t WS_TOTAL= WS_WIMG + (size_t)16 * LW_END * 2;

// weight-region tables (shared by init image builder and TDM loader)
constexpr int NREG = 14;
__device__ __constant__ const int REG_N[NREG]  = {256,256,768,768,768,768,768,768,256,256,256,256,256,64};
__device__ __constant__ const int REG_K[NREG]  = {520,256,384,384,384,256,256,256,256,256,256,1152,256,256};
__device__ __constant__ const int REG_LD[NREG] = {528,256,384,384,384,256,256,256,256,256,256,1152,256,256};
__device__ __constant__ const int REG_OF[NREG] = {LW_FW,LW_FWGLU,LW_WIH0,LW_WIH1,LW_WIH2,LW_WHH0,LW_WHH1,
                                                  LW_WHH2,LW_GLU0,LW_GLU1,LW_GLU2,LW_SKIP,LW_SKGLU,LW_OUT};

// ---------------- helpers ----------------------------------------------------
__device__ __forceinline__ int ownTiles(int NT, int g) {
  return (NT > g) ? ((NT - g + 15) >> 4) : 0;   // WG g owns global 16-col tiles {g, g+16, ...}
}
__device__ __forceinline__ float sigm(float x) { return 1.f / (1.f + __expf(-x)); }

// A fragment: 16x32 bf16, lane L -> M=L&15, kbase = k0 + 8*(L>=16)
// elements 0..7 -> K kbase..kbase+7 ; elements 8..15 -> K kbase+16..kbase+23  (ISA 7.12.2)
__device__ __forceinline__ v16bf load_a_frag(const __bf16* p) {
  BF16Frag f;
  f.q[0] = *(const u32x4*)(p);
  f.q[1] = *(const u32x4*)(p + 16);
  return f.v;
}
// B fragment: 32x16 bf16, lane L -> N=L&15, kbase = k0 + 8*(L>=16); elements 0..7 -> K kbase..kbase+7 ; 8..15 -> K kbase+16..kbase+23
__device__ __forceinline__ v16bf load_b_frag(const __bf16* p) {
  BF16Frag f;
  f.q[0] = *(const u32x4*)(p);
  f.q[1] = *(const u32x4*)(p + 16);
  return f.v;
}

// Y[64, nOwned*16] = X[64, Kp] @ W_rows^T, result staged to LDS fp32 Yl (ld=48).
template <int KP>
__device__ void wg_gemm(float* __restrict__ Yl,
                        const __bf16* __restrict__ X, int ldx,
                        const __bf16* __restrict__ W, int ldw,
                        int rowBase0, int tilePitchRows, int nOwned) {
  const int wave = threadIdx.x >> 5;
  const int lane = threadIdx.x & 31;
  const int mrow = lane & 15;
  const int hi   = lane >> 4;
  const int nTot = nOwned * 4;                 // 4 M-tiles (B=64)
  for (int t = wave; t < nTot; t += 8) {
    const int mt = t & 3;
    const int lt = t >> 2;
    const __bf16* xr = X + (size_t)(mt * 16 + mrow) * ldx + hi * 8;
    const __bf16* wr = W + (size_t)(rowBase0 + lt * tilePitchRows + mrow) * ldw + hi * 8;
    v8f acc = {};
#pragma unroll 8
    for (int k0 = 0; k0 < KP; k0 += 32) {
      v16bf a = load_a_frag(xr + k0);
      v16bf b = load_b_frag(wr + k0);
      acc = __builtin_amdgcn_wmma_f32_16x16x32_bf16(false, a, false, b,
                                                    (short)0, acc, false, false);
    }
    const int m0 = mt * 16 + hi * 8;           // D: lane<16 -> M=r, lane>=16 -> M=r+8
    const int c0 = lt * 16 + (lane & 15);
#pragma unroll
    for (int r = 0; r < 8; ++r) Yl[(m0 + r) * 48 + c0] = acc[r];
  }
}

// Tensor Data Mover: 2D tile (rows x ldw bf16, contiguous) global -> LDS.
// D# per CDNA5 ISA ch.8: group0 {count=1, lds_addr, global_addr, type=2},
// group1 {data_size=2B, tensor_dim0/1 = tile_dim0/1 = ldw/rows, stride0 = ldw}.
// This toolchain declares the 6-arg builtin: (g0, g1, g2, g3, g4, cpol).
__device__ __forceinline__ void tdm_load_2d(unsigned lds_byte, const void* gsrc,
                                            unsigned rows, unsigned ldw) {
  const unsigned long long ga = (unsigned long long)(size_t)gsrc;
  u32x4 g0;
  g0[0] = 1u;                                            // count=1 (valid user D#)
  g0[1] = lds_byte;                                      // lds_addr
  g0[2] = (unsigned)ga;                                  // global_addr[31:0]
  g0[3] = (unsigned)((ga >> 32) & 0x01FFFFFFu) | (2u << 30);  // addr[56:32] | type=2
  i32x8 g1;
  g1[0] = (int)(1u << 16);                               // data_size=1 -> 2 bytes
  g1[1] = (int)((ldw & 0xFFFFu) << 16);                  // tensor_dim0[15:0]
  g1[2] = (int)(((ldw >> 16) & 0xFFFFu) | ((rows & 0xFFFFu) << 16)); // t0 hi | t1 lo
  g1[3] = (int)(((rows >> 16) & 0xFFFFu) | ((ldw & 0xFFFFu) << 16)); // t1 hi | tile_dim0
  g1[4] = (int)(rows & 0xFFFFu);                         // tile_dim1 | tile_dim2=0
  g1[5] = (int)ldw;                                      // tensor_dim0_stride[31:0]
  g1[6] = 0;
  g1[7] = 0;
  i32x4 gz4 = {0, 0, 0, 0};                              // 2D: groups 2/3 unused
  i32x8 gz8 = {0, 0, 0, 0, 0, 0, 0, 0};
  __builtin_amdgcn_tensor_load_to_lds(g0, g1, gz4, gz4, gz8, 0);
}

// grid barrier: monotonic counter in L2 + cluster barrier (NOP if not clustered)
__device__ __forceinline__ void gsync(unsigned* cnt, unsigned& phase) {
  __syncthreads();
  if (threadIdx.x == 0) {
    __threadfence();
    atomicAdd(cnt, 1u);
    phase += 16u;
    while (atomicAdd(cnt, 0u) < phase) __builtin_amdgcn_s_sleep(2);
    __threadfence();
  }
  __builtin_amdgcn_s_cluster_barrier();
  __syncthreads();
}

// owned-slice tanh -> bf16 activation store (cond stages); global col = g*16 + lt*256 + nl
__device__ __forceinline__ void ew_tanh_bf16(const float* Yl, __bf16* dst, int ldd,
                                             int g, int nOwned) {
  const int per = nOwned * 16;
  for (int idx = threadIdx.x; idx < 64 * per; idx += 256) {
    const int b = idx / per, cl = idx - b * per;
    const int n = g * 16 + ((cl >> 4) << 8) + (cl & 15);
    dst[(size_t)b * ldd + n] = (__bf16)tanhf(Yl[b * 48 + cl]);
  }
}

// ---------------- init kernel ------------------------------------------------
// Converts cond weights to padded bf16, seeds ring/states, zeroes the barrier,
// and builds each WG's complete bf16 LDS weight image (shard layout) so the
// persistent kernel can DMA it with the TDM in one burst.
__global__ void fargan_init(const float* __restrict__ cw1, const float* __restrict__ cw2,
                            const float* __restrict__ cw3, const float* __restrict__ prev,
                            const float* __restrict__ fw_w, const float* __restrict__ fw_glu_w,
                            const float* __restrict__ g1ih, const float* __restrict__ g2ih,
                            const float* __restrict__ g3ih, const float* __restrict__ g1hh,
                            const float* __restrict__ g2hh, const float* __restrict__ g3hh,
                            const float* __restrict__ glu1, const float* __restrict__ glu2,
                            const float* __restrict__ glu3, const float* __restrict__ skip_w,
                            const float* __restrict__ skip_glu_w, const float* __restrict__ out_w,
                            char* __restrict__ ws) {
  const size_t tid = (size_t)blockIdx.x * blockDim.x + threadIdx.x;
  const size_t str = (size_t)gridDim.x * blockDim.x;
  if (tid == 0) *(unsigned*)(ws + WS_SYNC) = 0u;

  __bf16* c1 = (__bf16*)(ws + WS_CW1);
  __bf16* c2 = (__bf16*)(ws + WS_CW2);
  __bf16* c3 = (__bf16*)(ws + WS_CW3);
  for (size_t i = tid; i < (size_t)336 * KC; i += str) {
    const int n = (int)(i / KC), k = (int)(i % KC);
    c1[i] = (__bf16)((k < 336) ? cw1[n * 336 + k] : 0.f);
    c2[i] = (__bf16)((k < 336) ? cw2[n * 336 + k] : 0.f);
  }
  for (size_t i = tid; i < (size_t)512 * KC; i += str) {
    const int n = (int)(i / KC), k = (int)(i % KC);
    c3[i] = (__bf16)((k < 336) ? cw3[n * 336 + k] : 0.f);
  }
  float* ring = (float*)(ws + WS_RING);
  for (size_t i = tid; i < (size_t)B * 256; i += str) ring[i] = prev[i];
  __bf16* ac = (__bf16*)(ws + WS_ACIN);                  // acin/ac1/ac2 contiguous
  for (size_t i = tid; i < (size_t)3 * B * KC; i += str) ac[i] = (__bf16)0.f;
  __bf16* hb = (__bf16*)(ws + WS_H);
  for (size_t i = tid; i < (size_t)2 * 3 * B * 256; i += str) hb[i] = (__bf16)0.f;
  float* hf = (float*)(ws + WS_HF);
  for (size_t i = tid; i < (size_t)3 * B * 256; i += str) hf[i] = 0.f;
  float* s4 = (float*)(ws + WS_S4);
  for (size_t i = tid; i < (size_t)B * 264; i += str) s4[i] = 0.f;

  // per-WG sharded bf16 weight images
  const float* Wsrc[NREG] = {fw_w, fw_glu_w, g1ih, g2ih, g3ih, g1hh, g2hh, g3hh,
                             glu1, glu2, glu3, skip_w, skip_glu_w, out_w};
  __bf16* img = (__bf16*)(ws + WS_WIMG);
  for (int g = 0; g < 16; ++g) {
    __bf16* gi = img + (size_t)g * LW_END;
    for (int w = 0; w < NREG; ++w) {
      const int N = REG_N[w], K = REG_K[w], ldw = REG_LD[w], off = REG_OF[w];
      const int nOwn = (N / 16 > g) ? ((N / 16 - g + 15) >> 4) : 0;
      const int tot  = nOwn * 16 * ldw;
      const float* src = Wsrc[w];
      for (int idx = (int)tid; idx < tot; idx += (int)str) {
        const int row = idx / ldw, k = idx - row * ldw;
        const int lt = row >> 4, r = row & 15;
        const int n = (g + 16 * lt) * 16 + r;
        gi[off + idx] = (__bf16)((k < K) ? src[(size_t)n * K + k] : 0.f);
      }
    }
  }
}

// ---------------- persistent recurrence kernel -------------------------------
__global__ __launch_bounds__(256, 1)
void fargan_persistent(const float* __restrict__ features, const float* __restrict__ gfeat,
                       const int* __restrict__ periods,
                       const float* __restrict__ gain_w, const float* __restrict__ gain_b,
                       const float* __restrict__ pg_w,   const float* __restrict__ pg_b,
                       float* __restrict__ out, char* __restrict__ ws) {
  extern __shared__ __align__(16) char smem[];
  float*  Ylx = (float*)smem;
  float*  Ylh = Ylx + YL_ELEMS;
  __bf16* wls = (__bf16*)(smem + LDS_WBYTE);
  const int g = blockIdx.x;                    // 0..15

  // ---- workspace views ----
  unsigned* sync = (unsigned*)(ws + WS_SYNC);
  const __bf16* CW1 = (const __bf16*)(ws + WS_CW1);
  const __bf16* CW2 = (const __bf16*)(ws + WS_CW2);
  const __bf16* CW3 = (const __bf16*)(ws + WS_CW3);
  __bf16* ACIN = (__bf16*)(ws + WS_ACIN);
  __bf16* AC1  = (__bf16*)(ws + WS_AC1);
  __bf16* AC2  = (__bf16*)(ws + WS_AC2);
  __bf16* CONDB= (__bf16*)(ws + WS_COND);
  __bf16* X4   = (__bf16*)(ws + WS_X4);
  __bf16* FW   = (__bf16*)(ws + WS_FW);
  __bf16* SKb  = (__bf16*)(ws + WS_SK);
  __bf16* XG   = (__bf16*)(ws + WS_XG);
  __bf16* HB   = (__bf16*)(ws + WS_H);
  __bf16* SKX  = (__bf16*)(ws + WS_SKX);
  float*  hF   = (float*)(ws + WS_HF);
  float*  s4f  = (float*)(ws + WS_S4);
  float*  ring = (float*)(ws + WS_RING);
  float*  gainv= (float*)(ws + WS_GAIN);
  float*  pgv  = (float*)(ws + WS_PG);
  float*  plv  = (float*)(ws + WS_PL);
  float*  psub = (float*)(ws + WS_PSUB);

  // ---- one-time: TDM-stage this WG's weight image into LDS ----
  if ((threadIdx.x >> 5) == 0) {               // wave 0 queues the DMA descriptors
    const __bf16* img = (const __bf16*)(ws + WS_WIMG) + (size_t)g * LW_END;
    for (int w = 0; w < NREG; ++w) {
      const int rows = ownTiles(REG_N[w] >> 4, g) * 16;
      if (!rows) continue;
      tdm_load_2d((unsigned)(size_t)(wls + REG_OF[w]), img + REG_OF[w],
                  (unsigned)rows, (unsigned)REG_LD[w]);
    }
    __builtin_amdgcn_s_wait_tensorcnt(0);      // TENSORcnt == 0 -> shards resident
  }
  __syncthreads();

  unsigned phase = 0;
  const int nOwnC = ownTiles(21, g);           // cond N=336

  for (int f = 0; f < FRAMES; ++f) {
    // ---- S0: build cond input [feat | gf] (WG0) ----
    if (g == 0) {
      for (int idx = threadIdx.x; idx < B * 336; idx += 256) {
        const int b = idx / 336, c = idx - b * 336;
        const float v = (c < 80) ? features[(size_t)(b * 80 + c) * FRAMES + f]
                                 : gfeat[b * 256 + (c - 80)];
        ACIN[b * KC + c] = (__bf16)v;
      }
      if (threadIdx.x == 0 && f + 1 < FRAMES)
        __builtin_prefetch(features + (f + 1), 0, 1);   // global_prefetch_b8
    }
    gsync(sync, phase);

    // ---- S1..S3: cond MLP (weights streamed bf16 from L2; only 100x/call) ----
    wg_gemm<KC>(Ylx, ACIN, KC, CW1, KC, g * 16, 256, nOwnC);
    __syncthreads();  ew_tanh_bf16(Ylx, AC1, KC, g, nOwnC);  gsync(sync, phase);
    wg_gemm<KC>(Ylx, AC1, KC, CW2, KC, g * 16, 256, nOwnC);
    __syncthreads();  ew_tanh_bf16(Ylx, AC2, KC, g, nOwnC);  gsync(sync, phase);
    wg_gemm<KC>(Ylx, AC2, KC, CW3, KC, g * 16, 256, 2);
    __syncthreads();  ew_tanh_bf16(Ylx, CONDB, 512, g, 2);   gsync(sync, phase);

    for (int i = 0; i < 4; ++i) {
      const int t    = f * 4 + i;
      const int wpos = (t * 64) & 255;

      // ---- S4: gain, pitch gather, X4 = [sif | s4] (WG0, thread b) ----
      if (g == 0 && threadIdx.x < 64) {
        const int b   = threadIdx.x;
        const int per = periods[b * FRAMES + f];
        const __bf16* sf = CONDB + b * 512 + i * 128;
        float acc = gain_b[0];
        for (int k = 0; k < 128; ++k) acc += (float)sf[k] * gain_w[k];
        const float gn = __expf(acc);
        gainv[b] = gn;
        const float inv = 1.f / (1e-5f + gn);
        const float* rb = ring + b * 256;
        __bf16* x4 = X4 + b * 528;
        float*  s4 = s4f + b * 264;
        for (int k = 0; k < 128; ++k) x4[k] = sf[k];
        for (int n = 0; n < 64; ++n) {
          const float v = rb[(wpos + 192 + n) & 255] * inv;
          psub[b * 64 + n] = v;  x4[128 + n] = (__bf16)v;
        }
        for (int r = 0; r < 68; ++r) {
          int idx = 256 - per + r - 2;
          if (idx >= 256) idx -= per;
          const float v = rb[(wpos + idx) & 255] * inv;
          plv[b * 72 + r] = v;  x4[192 + r] = (__bf16)v;
        }
        for (int k = 0; k < 260; ++k) x4[260 + k] = (__bf16)s4[k];   // old s4
        for (int k = 520; k < 528; ++k) x4[k] = (__bf16)0.f;
        for (int k = 0; k < 128; ++k) s4[k] = (float)sf[k];          // new s4 = sif
        for (int n = 0; n < 64; ++n)  s4[128 + n] = psub[b * 64 + n];
        for (int r = 0; r < 68; ++r)  s4[192 + r] = plv[b * 72 + r];
      }
      gsync(sync, phase);

      // ---- S5: fw0 = tanh(X4 @ fw_w^T) ----
      wg_gemm<528>(Ylx, X4, 528, wls + LW_FW, 528, 0, 16, 1);
      __syncthreads();
      for (int idx = threadIdx.x; idx < 1024; idx += 256) {
        const int b = idx >> 4, nl = idx & 15, n = g * 16 + nl;
        FW[b * 256 + n] = (__bf16)tanhf(Ylx[b * 48 + nl]);
      }
      gsync(sync, phase);

      // ---- S6: fw = fw0 * sigm(fw0 @ fw_glu^T) -> XG0 head, SKX[768..1023] ----
      wg_gemm<256>(Ylx, FW, 256, wls + LW_FWGLU, 256, 0, 16, 1);
      __syncthreads();
      for (int idx = threadIdx.x; idx < 1024; idx += 256) {
        const int b = idx >> 4, nl = idx & 15, n = g * 16 + nl;
        const float v = (float)FW[b * 256 + n] * sigm(Ylx[b * 48 + nl]);
        XG[b * 384 + n]         = (__bf16)v;
        SKX[b * 1152 + 768 + n] = (__bf16)v;
      }
      gsync(sync, phase);

      // ---- S7: pitch gains pg, GRU/skip tails (WG0) ----
      if (g == 0) {
        {
          const int b = threadIdx.x >> 2, j = threadIdx.x & 3;
          const __bf16* fwv = XG + b * 384;
          float acc = pg_b[j];
          for (int k = 0; k < 256; ++k) acc += (float)fwv[k] * pg_w[j * 256 + k];
          pgv[b * 4 + j] = sigm(acc) + 1e-5f;
        }
        __syncthreads();
        if (threadIdx.x < 64) {
          const int b = threadIdx.x;
          for (int j = 0; j < 3; ++j) {
            const float p = pgv[b * 4 + j];
            __bf16* xg = XG + j * B * 384 + b * 384;
            for (int n = 0; n < 64; ++n) {
              xg[256 + n] = (__bf16)(p * plv[b * 72 + 2 + n]);
              xg[320 + n] = (__bf16)psub[b * 64 + n];
            }
          }
          const float p3 = pgv[b * 4 + 3];
          __bf16* sx = SKX + b * 1152;
          for (int n = 0; n < 64; ++n) {
            sx[1024 + n] = (__bf16)(p3 * plv[b * 72 + 2 + n]);
            sx[1088 + n] = (__bf16)psub[b * 64 + n];
          }
        }
      }
      gsync(sync, phase);

      // ---- 3x { GRU cell + GLU } ----
      for (int j = 0; j < 3; ++j) {
        const __bf16* hOld = HB + (size_t)(t & 1) * (3 * B * 256) + j * B * 256;
        __bf16*       hNew = HB + (size_t)((t + 1) & 1) * (3 * B * 256) + j * B * 256;
        wg_gemm<384>(Ylx, XG + j * B * 384, 384, wls + LW_WIH0 + j * (48 * 384), 384, 0, 16, 3);
        wg_gemm<256>(Ylh, hOld, 256,             wls + LW_WHH0 + j * (48 * 256), 256, 0, 16, 3);
        __syncthreads();
        // gate tiles for this WG: cols 0-15=r, 16-31=z, 32-47=n (N = {g,g+16,g+32}*16)
        for (int idx = threadIdx.x; idx < 1024; idx += 256) {
          const int b = idx >> 4, jl = idx & 15, col = g * 16 + jl;
          const float r = sigm(Ylx[b * 48 + jl]      + Ylh[b * 48 + jl]);
          const float z = sigm(Ylx[b * 48 + 16 + jl] + Ylh[b * 48 + 16 + jl]);
          const float n = tanhf(Ylx[b * 48 + 32 + jl] + r * Ylh[b * 48 + 32 + jl]);
          const float hold = hF[j * B * 256 + b * 256 + col];
          const float h = (1.f - z) * n + z * hold;
          hF[j * B * 256 + b * 256 + col] = h;
          hNew[b * 256 + col] = (__bf16)h;
        }
        gsync(sync, phase);

        // o_j = s_j * sigm(s_j @ glu_j^T)
        wg_gemm<256>(Ylx, hNew, 256, wls + LW_GLU0 + j * (16 * 256), 256, 0, 16, 1);
        __syncthreads();
        for (int idx = threadIdx.x; idx < 1024; idx += 256) {
          const int b = idx >> 4, nl = idx & 15, n = g * 16 + nl;
          const float v = hF[j * B * 256 + b * 256 + n] * sigm(Ylx[b * 48 + nl]);
          const __bf16 bv = (__bf16)v;
          if (j < 2) XG[(j + 1) * B * 384 + b * 384 + n] = bv;  // next GRU input head
          SKX[b * 1152 + j * 256 + n] = bv;                     // skip input slice
        }
        gsync(sync, phase);
      }

      // ---- S14: sk0 = tanh(SKX @ skip_w^T) ----
      wg_gemm<1152>(Ylx, SKX, 1152, wls + LW_SKIP, 1152, 0, 16, 1);
      __syncthreads();
      for (int idx = threadIdx.x; idx < 1024; idx += 256) {
        const int b = idx >> 4, nl = idx & 15, n = g * 16 + nl;
        FW[b * 256 + n] = (__bf16)tanhf(Ylx[b * 48 + nl]);
      }
      gsync(sync, phase);

      // ---- S15: sk = sk0 * sigm(sk0 @ skip_glu^T) ----
      wg_gemm<256>(Ylx, FW, 256, wls + LW_SKGLU, 256, 0, 16, 1);
      __syncthreads();
      for (int idx = threadIdx.x; idx < 1024; idx += 256) {
        const int b = idx >> 4, nl = idx & 15, n = g * 16 + nl;
        SKb[b * 256 + n] = (__bf16)((float)FW[b * 256 + n] * sigm(Ylx[b * 48 + nl]));
      }
      gsync(sync, phase);

      // ---- S16: out = tanh(sk @ out_w^T) * gain -> d_out and sample ring ----
      const int nOwnO = (g < 4) ? 1 : 0;
      wg_gemm<256>(Ylx, SKb, 256, wls + LW_OUT, 256, 0, 16, nOwnO);
      __syncthreads();
      if (nOwnO) {
        for (int idx = threadIdx.x; idx < 1024; idx += 256) {
          const int b = idx >> 4, nl = idx & 15, n = g * 16 + nl;
          const float v = tanhf(Ylx[b * 48 + nl]) * gainv[b];
          out[(size_t)b * (FRAMES * HOP) + f * HOP + i * SUB + n] = v;
          ring[b * 256 + ((wpos + n) & 255)] = v;
        }
      }
      gsync(sync, phase);
    }
  }
}

// ---------------- host entry -------------------------------------------------
extern "C" void kernel_launch(void* const* d_in, const int* in_sizes, int n_in,
                              void* d_out, int out_size, void* d_ws, size_t ws_size,
                              hipStream_t stream) {
  (void)in_sizes; (void)n_in; (void)out_size; (void)ws_size;
  const float* features = (const float*)d_in[0];
  const float* gfeat    = (const float*)d_in[1];
  const float* prev     = (const float*)d_in[2];
  const int*   periods  = (const int*)d_in[3];
  const float* cond_w1  = (const float*)d_in[4];
  const float* cond_w2  = (const float*)d_in[5];
  const float* cond_w3  = (const float*)d_in[6];
  const float* gain_w   = (const float*)d_in[7];
  const float* gain_b   = (const float*)d_in[8];
  const float* fw_w     = (const float*)d_in[9];
  const float* fw_glu_w = (const float*)d_in[10];
  const float* pg_w     = (const float*)d_in[11];
  const float* pg_b     = (const float*)d_in[12];
  const float* g1ih     = (const float*)d_in[13];
  const float* g1hh     = (const float*)d_in[14];
  const float* g2ih     = (const float*)d_in[15];
  const float* g2hh     = (const float*)d_in[16];
  const float* g3ih     = (const float*)d_in[17];
  const float* g3hh     = (const float*)d_in[18];
  const float* glu1     = (const float*)d_in[19];
  const float* glu2     = (const float*)d_in[20];
  const float* glu3     = (const float*)d_in[21];
  const float* skip_w   = (const float*)d_in[22];
  const float* skip_glu = (const float*)d_in[23];
  const float* out_w    = (const float*)d_in[24];

  (void)hipFuncSetAttribute((const void*)fargan_persistent,
                            hipFuncAttributeMaxDynamicSharedMemorySize, LDS_BYTES);

  hipLaunchKernelGGL(fargan_init, dim3(512), dim3(256), 0, stream,
                     cond_w1, cond_w2, cond_w3, prev,
                     fw_w, fw_glu_w, g1ih, g2ih, g3ih, g1hh, g2hh, g3hh,
                     glu1, glu2, glu3, skip_w, skip_glu, out_w, (char*)d_ws);
  hipLaunchKernelGGL(fargan_persistent, dim3(16), dim3(256), LDS_BYTES, stream,
                     features, gfeat, periods, gain_w, gain_b, pg_w, pg_b,
                     (float*)d_out, (char*)d_ws);
}
